// STAGATE_30520037605630
// MI455X (gfx1250) — matrix operand order repaired and checked by reading, not compiled, not run on hardware
//
#include <hip/hip_runtime.h>
#include <hip/hip_bf16.h>

// ---------------------------------------------------------------------------
// STAGATE forward for MI455X (gfx1250, wave32).
// All four GEMMs run through v_wmma_f32_16x16x32_f16 (f16 in, f32 accum).
// K is zero-padded to a multiple of 32 at staging time so the WMMA inner loop
// has NO guards: just global_load_b128 + v_wmma. Addressing uses uniform SGPR
// bases + unsigned 32-bit lane offsets (no sign-extends, no spills).
// Edge softmax/scatter passes are atomics-based; x_proj (82MB) fits in the
// 192MB L2 so the gather/scatter traffic stays on-chip.
// ---------------------------------------------------------------------------

typedef _Float16 v8h  __attribute__((ext_vector_type(8)));
typedef _Float16 v16h __attribute__((ext_vector_type(16)));
typedef float    v8f  __attribute__((ext_vector_type(8)));

#define NNODE  40000
#define IN_DIM 2000
#define KP1    2016      // IN_DIM padded to multiple of 32
#define HID    512
#define OUTD   30
#define NEDGE  400000

// ---------------------------------------------------------------------------
// C[M x Nn] f32 (row major, ldc) = A[M x K] f16 (row major, lda)
//                                  * BT[Nn x K]^T f16 (row major, ldb)
// 8 waves / block; each wave computes an independent 32x64 tile (2x4 WMMA).
// Requirements: M % 32 == 0, K % 32 == 0 (zero-padded), lda/ldb >= K,
//               all arrays < 2^31 elements (32-bit lane offsets).
//
// Fragment layouts per CDNA5 ISA 7.12.2 (wave32):
//   A 16x32 f16 : lane L -> row M=L%16; halves 0..7 at K = k0 + (L/16)*8,
//                 halves 8..15 at K = k0 + 16 + (L/16)*8.
//   B 32x16 f16 : lane L -> col N=L%16; halves 0..7 at K = k0 + (L/16)*8,
//                 halves 8..15 at K = k0 + 16 + (L/16)*8 (same order as A).
//   C 16x16 f32 : VGPR v, lane L -> (M = v + 8*(L/16), N = L%16).
// ---------------------------------------------------------------------------
__global__ void __launch_bounds__(256)
gemm_wmma_f16(const _Float16* __restrict__ A, int lda,
              const _Float16* __restrict__ BT, int ldb,
              float* __restrict__ C, int ldc,
              int Nn, int K, int Mt, int Nt) {
  int t = blockIdx.x * 8 + (threadIdx.x >> 5);   // global wave -> output tile
  if (t >= Mt * Nt) return;                      // wave-uniform: EXEC stays full
  int tm = t % Mt, tn = t / Mt;                  // column-major over M for BT reuse
  int lane = threadIdx.x & 31;
  int r = lane & 15, hi = lane >> 4;
  int m0 = tm * 32, n0 = tn * 64;

  // unsigned 32-bit element offsets; uniform SGPR base + lane voffset addressing
  unsigned aoff0 = (unsigned)((m0 + r) * lda + hi * 8);
  unsigned aoff1 = aoff0 + (unsigned)(16 * lda);
  unsigned boff[4];
  int cn[4];
#pragma unroll
  for (int j = 0; j < 4; ++j) {
    int n = n0 + j * 16 + r;
    cn[j] = n;
    int nc = (n < Nn) ? n : (Nn - 1);            // clamp: compute garbage, never store
    boff[j] = (unsigned)(nc * ldb + hi * 8);
  }

  v8f acc[2][4] = {};

  for (unsigned k0 = 0; k0 < (unsigned)K; k0 += 32) {
    union { v16h v; v8h h[2]; } af[2], bf[4];
    af[0].h[0] = *(const v8h*)(A + (size_t)(aoff0 + k0));
    af[0].h[1] = *(const v8h*)(A + (size_t)(aoff0 + k0 + 16));
    af[1].h[0] = *(const v8h*)(A + (size_t)(aoff1 + k0));
    af[1].h[1] = *(const v8h*)(A + (size_t)(aoff1 + k0 + 16));
#pragma unroll
    for (int j = 0; j < 4; ++j) {
      bf[j].h[0] = *(const v8h*)(BT + (size_t)(boff[j] + k0));
      bf[j].h[1] = *(const v8h*)(BT + (size_t)(boff[j] + k0 + 16));
    }

    if (k0 + 128 < (unsigned)K) {       // -> global_prefetch_b8, uniform branch
      __builtin_prefetch(A + (size_t)(aoff0 + k0 + 128), 0, 1);
      __builtin_prefetch(A + (size_t)(aoff1 + k0 + 128), 0, 1);
    }

#pragma unroll
    for (int i = 0; i < 2; ++i)
#pragma unroll
      for (int j = 0; j < 4; ++j)
        acc[i][j] = __builtin_amdgcn_wmma_f32_16x16x32_f16(
            false, af[i].v, false, bf[j].v, (short)0, acc[i][j], false, false);
  }

  // store: lanes r=0..15 write 16 consecutive dwords per row -> coalesced
#pragma unroll
  for (int i = 0; i < 2; ++i) {
    float* rowb = C + (size_t)(m0 + i * 16 + 8 * hi) * ldc;
#pragma unroll
    for (int j = 0; j < 4; ++j) {
      if (cn[j] < Nn) {
#pragma unroll
        for (int v = 0; v < 8; ++v) rowb[(size_t)v * ldc + cn[j]] = acc[i][j][v];
      }
    }
  }
}

// ------------------------------ helper kernels -----------------------------

__global__ void zero_f32(float* p, long n) {
  long t = (long)blockIdx.x * blockDim.x + threadIdx.x;
  if (t < n) p[t] = 0.0f;
}

// f32 [R x C] -> f16 [R x Cp] with zero K-padding (C == Cp means plain convert)
__global__ void cvt_pad(const float* __restrict__ in, _Float16* __restrict__ out,
                        int R, int Cc, int Cp) {
  long t = (long)blockIdx.x * blockDim.x + threadIdx.x;
  if (t >= (long)R * Cp) return;
  int r = (int)(t / Cp), c = (int)(t % Cp);
  out[t] = (c < Cc) ? (_Float16)in[(size_t)r * Cc + c] : (_Float16)0;
}

// in f32 [R x C] -> out f16 [C x Rp], out[c][k] = (k<R) ? in[k][c] : 0
__global__ void transpose_cvt_pad(const float* __restrict__ in, _Float16* __restrict__ out,
                                  int R, int Cc, int Rp) {
  long t = (long)blockIdx.x * blockDim.x + threadIdx.x;
  if (t >= (long)Cc * Rp) return;
  int c = (int)(t / Rp), k = (int)(t % Rp);
  out[t] = (k < R) ? (_Float16)in[(size_t)k * Cc + c] : (_Float16)0;
}

// W2 [HID x OUTD] -> w2a [HID x 32] (K-padded), w2b [32 x HID] (N-padded transpose)
__global__ void w2_stage(const float* __restrict__ W2, _Float16* __restrict__ w2a,
                         _Float16* __restrict__ w2b) {
  int t = blockIdx.x * blockDim.x + threadIdx.x;
  if (t >= HID * 32) return;
  int n = t >> 5, k = t & 31;
  float v = (k < OUTD) ? W2[(size_t)n * OUTD + k] : 0.0f;
  w2a[t] = (_Float16)v;                   // w2a[n][k] = W2[n][k]
  w2b[(size_t)k * HID + n] = (_Float16)v; // w2b[k][n] = W2[n][k]  (rows >=30 zero)
}

// per-node attention scores: one wave per node, 512-wide dot with shuffle reduce
__global__ void att_dots(const float* __restrict__ x1, const float* __restrict__ att_s,
                         const float* __restrict__ att_d, float* __restrict__ as,
                         float* __restrict__ ad) {
  int w = (blockIdx.x * blockDim.x + threadIdx.x) >> 5;
  int lane = threadIdx.x & 31;
  if (w >= NNODE) return;
  const float* row = x1 + (size_t)w * HID;
  float s = 0.f, d = 0.f;
  for (int c = lane; c < HID; c += 32) {
    float v = row[c];
    s += v * att_s[c];
    d += v * att_d[c];
  }
  for (int off = 16; off; off >>= 1) {
    s += __shfl_xor(s, off);
    d += __shfl_xor(d, off);
  }
  if (lane == 0) { as[w] = s; ad[w] = d; }
}

// score[e] = sigmoid(as[src]+ad[dst]); segmax via uint atomicMax (scores > 0)
__global__ void edge_score(const int* __restrict__ src, const int* __restrict__ dst,
                           const float* __restrict__ as, const float* __restrict__ ad,
                           float* __restrict__ score, unsigned* __restrict__ segmax) {
  int e = blockIdx.x * blockDim.x + threadIdx.x;
  if (e >= NEDGE) return;
  float x = as[src[e]] + ad[dst[e]];
  float s = 1.0f / (1.0f + expf(-x));
  score[e] = s;
  atomicMax(segmax + dst[e], __float_as_uint(s));  // positive floats: bit order == value order
}

__global__ void edge_exp(const int* __restrict__ dst, const unsigned* __restrict__ segmax,
                         float* __restrict__ score_ev, float* __restrict__ segsum) {
  int e = blockIdx.x * blockDim.x + threadIdx.x;
  if (e >= NEDGE) return;
  float m = __uint_as_float(segmax[dst[e]]);
  float ev = expf(score_ev[e] - m);
  score_ev[e] = ev;
  atomicAdd(segsum + dst[e], ev);
}

__global__ void edge_alpha(const int* __restrict__ dst, const float* __restrict__ segsum,
                           float* __restrict__ ev_alpha) {
  int e = blockIdx.x * blockDim.x + threadIdx.x;
  if (e >= NEDGE) return;
  ev_alpha[e] = ev_alpha[e] / (segsum[dst[e]] + 1e-16f);
}

// agg[dst,:] += x_proj[src,:]*alpha[e]; 128 threads/edge, float4 per thread
__global__ void scatter_msg(const int* __restrict__ src, const int* __restrict__ dst,
                            const float* __restrict__ alpha, const float* __restrict__ xp,
                            float* __restrict__ agg) {
  long tid = (long)blockIdx.x * blockDim.x + threadIdx.x;
  if (tid >= (long)NEDGE * 128) return;
  int e  = (int)(tid >> 7);
  int c4 = (int)(tid & 127) << 2;
  float a = alpha[e];
  const float4 v = *(const float4*)(xp + (size_t)src[e] * HID + c4);
  float* o = agg + (size_t)dst[e] * HID + c4;
  atomicAdd(o + 0, v.x * a);
  atomicAdd(o + 1, v.y * a);
  atomicAdd(o + 2, v.z * a);
  atomicAdd(o + 3, v.w * a);
}

__global__ void elu_cvt(const float* __restrict__ in, _Float16* __restrict__ out, long n) {
  long t = (long)blockIdx.x * blockDim.x + threadIdx.x;
  if (t >= n) return;
  float x = in[t];
  out[t] = (_Float16)(x > 0.0f ? x : expm1f(x));
}

// h2f [N x 32] -> d_out h2 [N x 30] f32 + h2_16 [N x 32] f16 (cols 30,31 are 0)
__global__ void h2_finalize(const float* __restrict__ h2f, float* __restrict__ out30,
                            _Float16* __restrict__ h2_16) {
  long t = (long)blockIdx.x * blockDim.x + threadIdx.x;
  if (t >= (long)NNODE * 32) return;
  int r = (int)(t >> 5), c = (int)(t & 31);
  float v = h2f[t];
  h2_16[t] = (_Float16)v;
  if (c < OUTD) out30[(size_t)r * OUTD + c] = v;
}

// ---------------------------------------------------------------------------

static inline size_t alignup(size_t x) { return (x + 255) & ~(size_t)255; }

extern "C" void kernel_launch(void* const* d_in, const int* in_sizes, int n_in,
                              void* d_out, int out_size, void* d_ws, size_t ws_size,
                              hipStream_t stream) {
  (void)in_sizes; (void)n_in; (void)out_size; (void)ws_size;

  const float* features = (const float*)d_in[0];   // [N, IN_DIM]
  const float* W1       = (const float*)d_in[1];   // [IN_DIM, HID]
  const float* W2       = (const float*)d_in[2];   // [HID, OUTD]
  const float* att_src  = (const float*)d_in[3];   // [HID]
  const float* att_dst  = (const float*)d_in[4];   // [HID]
  const int*   ei       = (const int*)d_in[5];     // [2, E]
  const int*   src = ei;
  const int*   dst = ei + NEDGE;

  float* out_h2 = (float*)d_out;                        // [N, OUTD]
  float* out_h4 = (float*)d_out + (size_t)NNODE * OUTD; // [N, IN_DIM]

  // ---- bump-allocate workspace (~382 MB total) ----
  char* ws = (char*)d_ws;
  size_t off = 0;
  auto take = [&](size_t bytes) { char* p = ws + off; off += alignup(bytes); return p; };

  _Float16* feat16 = (_Float16*)take((size_t)NNODE * KP1 * 2);    // K-padded A, GEMM1
  _Float16* w1_16  = (_Float16*)take((size_t)IN_DIM * HID * 2);   // BT for GEMM4
  _Float16* w1t16  = (_Float16*)take((size_t)HID * KP1 * 2);      // K-padded BT, GEMM1
  _Float16* w2a    = (_Float16*)take((size_t)HID * 32 * 2);       // BT for GEMM3
  _Float16* w2b    = (_Float16*)take((size_t)32 * HID * 2);       // BT for GEMM2
  float*    x1     = (float*)take((size_t)NNODE * HID * 4);       // reused as x3
  float*    a_s    = (float*)take((size_t)NNODE * 4);
  float*    a_d    = (float*)take((size_t)NNODE * 4);
  unsigned* segmax = (unsigned*)take((size_t)NNODE * 4);
  float*    segsum = (float*)take((size_t)NNODE * 4);
  float*    alpha  = (float*)take((size_t)NEDGE * 4);             // score -> e -> alpha
  float*    agg    = (float*)take((size_t)NNODE * HID * 4);
  _Float16* h16    = (_Float16*)take((size_t)NNODE * HID * 2);    // h1_16 then h3_16
  float*    h2f    = (float*)take((size_t)NNODE * 32 * 4);
  _Float16* h2_16  = (_Float16*)take((size_t)NNODE * 32 * 2);

  const int B = 256;
  auto blocks = [](long n) { return (unsigned)((n + 255) / 256); };
  auto gemm = [&](const _Float16* A, int lda, const _Float16* BT, int ldb,
                  float* C, int ldc, int Nn, int K) {
    int Mt = NNODE / 32;
    int Nt = (Nn + 63) / 64;
    long waves = (long)Mt * Nt;
    gemm_wmma_f16<<<(unsigned)((waves + 7) / 8), B, 0, stream>>>(A, lda, BT, ldb, C, ldc,
                                                                 Nn, K, Mt, Nt);
  };

  // ---- stage f16 operands (K-padded where needed) ----
  cvt_pad<<<blocks((long)NNODE * KP1), B, 0, stream>>>(features, feat16, NNODE, IN_DIM, KP1);
  cvt_pad<<<blocks((long)IN_DIM * HID), B, 0, stream>>>(W1, w1_16, IN_DIM, HID, HID);
  transpose_cvt_pad<<<blocks((long)HID * KP1), B, 0, stream>>>(W1, w1t16, IN_DIM, HID, KP1);
  w2_stage<<<blocks(HID * 32), B, 0, stream>>>(W2, w2a, w2b);

  // ---- conv1: x1 = feat @ W1 ----
  gemm(feat16, KP1, w1t16, KP1, x1, HID, HID, KP1);

  // ---- attention scores (computed once, reused by conv3) ----
  att_dots<<<blocks((long)NNODE * 32), B, 0, stream>>>(x1, att_src, att_dst, a_s, a_d);

  zero_f32<<<blocks(NNODE), B, 0, stream>>>((float*)segmax, NNODE);
  zero_f32<<<blocks(NNODE), B, 0, stream>>>(segsum, NNODE);
  edge_score<<<blocks(NEDGE), B, 0, stream>>>(src, dst, a_s, a_d, alpha, segmax);
  edge_exp<<<blocks(NEDGE), B, 0, stream>>>(dst, segmax, alpha, segsum);
  edge_alpha<<<blocks(NEDGE), B, 0, stream>>>(dst, segsum, alpha);

  // ---- propagate 1: agg = scatter_add(x1[src]*alpha) ; h1 = elu ----
  zero_f32<<<blocks((long)NNODE * HID), B, 0, stream>>>(agg, (long)NNODE * HID);
  scatter_msg<<<blocks((long)NEDGE * 128), B, 0, stream>>>(src, dst, alpha, x1, agg);
  elu_cvt<<<blocks((long)NNODE * HID), B, 0, stream>>>(agg, h16, (long)NNODE * HID);

  // ---- conv2: h2 = h1 @ W2 (Nn padded to 32) ----
  gemm(h16, HID, w2b, HID, h2f, 32, 32, HID);
  h2_finalize<<<blocks((long)NNODE * 32), B, 0, stream>>>(h2f, out_h2, h2_16);

  // ---- conv3 projection: x3 = h2 @ W2^T (K padded to 32); reuse x1 buffer ----
  gemm(h2_16, 32, w2a, 32, x1, HID, HID, 32);

  // ---- propagate 2 with tied alphas ; h3 = elu ----
  zero_f32<<<blocks((long)NNODE * HID), B, 0, stream>>>(agg, (long)NNODE * HID);
  scatter_msg<<<blocks((long)NEDGE * 128), B, 0, stream>>>(src, dst, alpha, x1, agg);
  elu_cvt<<<blocks((long)NNODE * HID), B, 0, stream>>>(agg, h16, (long)NNODE * HID);

  // ---- conv4: h4 = h3 @ W1^T (straight into d_out) ----
  gemm(h16, HID, w1_16, HID, out_h4, IN_DIM, IN_DIM, HID);
}
